// ActorAgent_27625229647898
// MI455X (gfx1250) — matrix operand order repaired and checked by reading, not compiled, not run on hardware
//
#include <hip/hip_runtime.h>
#include <hip/hip_bf16.h>
#include <math.h>

// ---------------- problem dims ----------------
#define T_DIM   512
#define B_SZ    8
#define OBS_D   256
#define D_INNER 2048
#define H_HEADS 16
#define N_STATE 64
#define P_HEAD  128
#define N_UNITS 256
#define N_ACT   64
#define M_TOT   (T_DIM * B_SZ)   // 4096

typedef __bf16 bf16_t;
typedef __attribute__((ext_vector_type(16))) __bf16 v16bf;
typedef __attribute__((ext_vector_type(8)))  __bf16 v8bf;
typedef __attribute__((ext_vector_type(8)))  float  v8f;

#if defined(__HIP_DEVICE_COMPILE__) && \
    __has_builtin(__builtin_amdgcn_global_load_async_to_lds_b32) && \
    __has_builtin(__builtin_amdgcn_s_wait_asynccnt)
#define USE_ASYNC_LDS 1
typedef __attribute__((address_space(1))) int gint_as;  // "__device__ int"
typedef __attribute__((address_space(3))) int lint_as;  // "__shared__ int"
#else
#define USE_ASYNC_LDS 0
#endif

// ---------------- prep: fp32 -> bf16 plain convert ----------------
__global__ __launch_bounds__(256) void cvt_bf16_kernel(
    const float* __restrict__ in, bf16_t* __restrict__ out, int n) {
  int i = blockIdx.x * 256 + threadIdx.x;
  if (i < n) out[i] = (bf16_t)in[i];
}

// ---------------- prep: fp32 KxN -> bf16 NxK (transpose convert) --------
__global__ __launch_bounds__(256) void tconv_kernel(
    const float* __restrict__ in, bf16_t* __restrict__ out, int Kd, int Nd) {
  int idx = blockIdx.x * 256 + threadIdx.x;
  if (idx >= Kd * Nd) return;
  int n = idx / Kd;
  int k = idx - n * Kd;
  out[idx] = (bf16_t)in[(size_t)k * Nd + n];   // out[n*Kd + k] = in[k][n]
}

// ---------------- WMMA fragment loaders (ISA 7.12.2 layouts) ------------
// A (16x32 bf16, MxK): lanes 0-15 row M=lane, K = {k..k+7, k+16..k+23};
//                      lanes 16-31 same rows, K = {k+8..k+15, k+24..k+31}.
__device__ __forceinline__ v16bf load_a_frag(const bf16_t* __restrict__ rowp,
                                             int k, int khalf) {
  const bf16_t* base = rowp + k + khalf;
  v8bf lo = *(const v8bf*)(base);
  v8bf hi = *(const v8bf*)(base + 16);
  v16bf a;
#pragma unroll
  for (int i = 0; i < 8; ++i) { a[i] = lo[i]; a[i + 8] = hi[i]; }
  return a;
}

// B (32x16 bf16, KxN): column N = lane&15; lanes 0-15 hold K={k..k+7, k+16..k+23},
// lanes 16-31 hold K={k+8..k+15, k+24..k+31}. Weights stored transposed (Wt[n][k]) so a
// column of W is contiguous: two 16B loads.
__device__ __forceinline__ v16bf load_b_frag(const bf16_t* __restrict__ colp,
                                             int k, int kboff) {
  const bf16_t* base = colp + k + kboff;
  v8bf lo = *(const v8bf*)(base);
  v8bf hi = *(const v8bf*)(base + 16);
  v16bf b;
#pragma unroll
  for (int i = 0; i < 8; ++i) { b[i] = lo[i]; b[i + 8] = hi[i]; }
  return b;
}

// ---------------- generic WMMA GEMM: out = epilogue(A @ Wt^T) -----------
// A: M x K bf16 row-major. Wt: N x K bf16 (W transposed). Requires K%64==0.
// Block: 256 threads = 8 waves, 4(M) x 2(N); wave tile 16 x (16*NSUB).
// All tile indices go through readfirstlane so guards are scalar branches
// and EXEC stays all-1s through every v_wmma (ISA 7.12 requirement).
// MODE: 0/3 bf16 relu(acc+bias) | 1 bf16 acc | 2 softplus+decay | 4 f32+bias
template <int MODE, int NSUB>
__global__ __launch_bounds__(256) void wmma_gemm_kernel(
    const bf16_t* __restrict__ A, const bf16_t* __restrict__ Wt,
    const float* __restrict__ bias, const float* __restrict__ aux,
    void* __restrict__ out, float* __restrict__ out2,
    int M, int N, int K) {
  const int tid  = threadIdx.x;
  const int lane = tid & 31;
  const int w    = __builtin_amdgcn_readfirstlane(tid >> 5);  // scalar wave id
  const int wm   = w & 3;
  const int wn   = w >> 2;
  const int m0   = blockIdx.x * 64 + wm * 16;          // scalar
  const int n0   = blockIdx.y * 128 + wn * 64;         // scalar
  if (n0 >= N) return;                                 // scalar branch

  const int khalf = (lane < 16) ? 0 : 8;   // A-frag K-half select
  const int kboff = (lane < 16) ? 0 : 8;   // B-frag K-half select
  const int ncol  = lane & 15;

  const bf16_t* aptr = A + (size_t)(m0 + ncol) * K;
  const bf16_t* bptr[NSUB];
#pragma unroll
  for (int j = 0; j < NSUB; ++j)
    bptr[j] = Wt + (size_t)(n0 + j * 16 + ncol) * K;

  v8f acc[NSUB];
#pragma unroll
  for (int j = 0; j < NSUB; ++j)
#pragma unroll
    for (int i = 0; i < 8; ++i) acc[j][i] = 0.0f;

  // Software-pipelined K loop, unrolled by 2 (K % 64 == 0): loads for the
  // next 32-wide slice are issued before the WMMA on the current slice.
  v16bf a0 = load_a_frag(aptr, 0, khalf);
  v16bf b0[NSUB];
#pragma unroll
  for (int j = 0; j < NSUB; ++j) b0[j] = load_b_frag(bptr[j], 0, kboff);

  for (int k = 0; k < K; k += 64) {
    if (k + 128 < K) {
      __builtin_prefetch(aptr + k + 128, 0, 1);
      __builtin_prefetch(bptr[0] + k + 128, 0, 1);
    }
    v16bf a1 = load_a_frag(aptr, k + 32, khalf);
    v16bf b1[NSUB];
#pragma unroll
    for (int j = 0; j < NSUB; ++j) b1[j] = load_b_frag(bptr[j], k + 32, kboff);
#pragma unroll
    for (int j = 0; j < NSUB; ++j)
      acc[j] = __builtin_amdgcn_wmma_f32_16x16x32_bf16(
          false, a0, false, b0[j], (short)0, acc[j], false, false);
    if (k + 64 < K) {
      a0 = load_a_frag(aptr, k + 64, khalf);
#pragma unroll
      for (int j = 0; j < NSUB; ++j) b0[j] = load_b_frag(bptr[j], k + 64, kboff);
    }
#pragma unroll
    for (int j = 0; j < NSUB; ++j)
      acc[j] = __builtin_amdgcn_wmma_f32_16x16x32_bf16(
          false, a1, false, b1[j], (short)0, acc[j], false, false);
  }

  // Epilogue. C/D layout: VGPR v -> row m0 + v + (lane<16?0:8), col lane&15.
  const int rowOff = (lane < 16) ? 0 : 8;
#pragma unroll
  for (int j = 0; j < NSUB; ++j) {
    const int col = n0 + j * 16 + ncol;
    const float bv = (MODE == 1) ? 0.0f : bias[col];
#pragma unroll
    for (int v = 0; v < 8; ++v) {
      const int row = m0 + rowOff + v;
      float val = acc[j][v] + bv;
      const size_t o = (size_t)row * N + col;
      if (MODE == 0 || MODE == 3) {
        val = val > 0.0f ? val : 0.0f;
        ((bf16_t*)out)[o] = (bf16_t)val;
      } else if (MODE == 1) {
        ((bf16_t*)out)[o] = (bf16_t)val;
      } else if (MODE == 2) {
        const float sp = (val > 20.0f) ? val : log1pf(expf(val));
        ((float*)out)[o] = sp;
        out2[o] = expf(-expf(aux[col]) * sp);
      } else {
        ((float*)out)[o] = val;
      }
    }
  }
}

// ---------------- SSD sequential scan -----------------------------------
// One block per (batch b, head h): 128 blocks, 128 threads (4 waves).
// Thread p owns state row h[p][0..63] in 64 VGPRs. Per step:
//   h[p][n] = dec*h[p][n] + (dt*x[p])*B[n];  y[p] = sum_n h[p][n]*C[n]
// B/C (64 bf16 each = 64 dwords) staged to LDS per step via the gfx1250
// async global->LDS path when available (ASYNCcnt + s_wait_asynccnt).
__device__ __forceinline__ float bflo(uint32_t w) {
  return __uint_as_float(w << 16);
}
__device__ __forceinline__ float bfhi(uint32_t w) {
  return __uint_as_float(w & 0xffff0000u);
}

__global__ __launch_bounds__(128) void ssd_scan_kernel(
    const bf16_t* __restrict__ X,   // M_TOT x D_INNER
    const float*  __restrict__ dtp, // M_TOT x H
    const float*  __restrict__ dec, // M_TOT x H
    const bf16_t* __restrict__ Bm,  // M_TOT x (H*N_STATE)
    const bf16_t* __restrict__ Cm,  // M_TOT x (H*N_STATE)
    bf16_t* __restrict__ Y) {       // M_TOT x D_INNER
  const int bh = blockIdx.x;
  const int b  = bh >> 4;          // 0..7
  const int h  = bh & 15;          // 0..15
  const int p  = threadIdx.x;      // 0..127

  // stage[0..31]  = B as 32 dwords (2 bf16 each)
  // stage[32..63] = C as 32 dwords
  __shared__ __align__(16) uint32_t stage[2 * N_STATE / 2];
  __shared__ float sc[2];

  float hst[N_STATE];
#pragma unroll
  for (int n = 0; n < N_STATE; ++n) hst[n] = 0.0f;

  for (int t = 0; t < T_DIM; ++t) {
    const int r = t * B_SZ + b;
    const size_t bcOff = (size_t)r * (H_HEADS * N_STATE) + h * N_STATE;
    if (p < 64) {
      const uint32_t* gsrc =
          (p < 32) ? (const uint32_t*)(Bm + bcOff) + p
                   : (const uint32_t*)(Cm + bcOff) + (p - 32);
#if USE_ASYNC_LDS
      __builtin_amdgcn_global_load_async_to_lds_b32(
          (gint_as*)gsrc, (lint_as*)&stage[p], 0, 0);
#else
      stage[p] = *gsrc;
#endif
    }
    if (p == 0) {
      sc[0] = dtp[r * H_HEADS + h];
      sc[1] = dec[r * H_HEADS + h];
    }
#if USE_ASYNC_LDS
    __builtin_amdgcn_s_wait_asynccnt(0);
#endif
    __syncthreads();

    const size_t xo = (size_t)r * D_INNER + h * P_HEAD + p;
    const float xv = (float)X[xo];
    const float c  = sc[0] * xv;
    const float d  = sc[1];
    float y = 0.0f;
    const uint4* S4 = (const uint4*)stage;
#pragma unroll
    for (int q = 0; q < 8; ++q) {
      const uint4 wb = S4[q];       // B dwords 4q..4q+3 (8 bf16)
      const uint4 wc = S4[8 + q];   // C dwords
      const uint32_t bw[4] = {wb.x, wb.y, wb.z, wb.w};
      const uint32_t cw[4] = {wc.x, wc.y, wc.z, wc.w};
#pragma unroll
      for (int i = 0; i < 4; ++i) {
        const int n = q * 8 + i * 2;
        const float bn0 = bflo(bw[i]), bn1 = bfhi(bw[i]);
        const float cn0 = bflo(cw[i]), cn1 = bfhi(cw[i]);
        hst[n]     = fmaf(d, hst[n],     c * bn0); y = fmaf(hst[n],     cn0, y);
        hst[n + 1] = fmaf(d, hst[n + 1], c * bn1); y = fmaf(hst[n + 1], cn1, y);
      }
    }
    Y[xo] = (bf16_t)y;
    __syncthreads();   // protect stage before next step's overwrite
  }
}

// ---------------- host-side orchestration -------------------------------
extern "C" void kernel_launch(void* const* d_in, const int* in_sizes, int n_in,
                              void* d_out, int out_size, void* d_ws, size_t ws_size,
                              hipStream_t stream) {
  const float* obs     = (const float*)d_in[0];
  const float* W_in    = (const float*)d_in[1];
  const float* b_in    = (const float*)d_in[2];
  const float* A_log   = (const float*)d_in[3];
  const float* dt_bias = (const float*)d_in[4];
  const float* W_dt    = (const float*)d_in[5];
  const float* W_B     = (const float*)d_in[6];
  const float* W_C     = (const float*)d_in[7];
  const float* W_yo    = (const float*)d_in[8];
  const float* b_yo    = (const float*)d_in[9];
  const float* W_head  = (const float*)d_in[10];
  const float* b_head  = (const float*)d_in[11];

  char* ws = (char*)d_ws;
  size_t off = 0;
  auto carve = [&](size_t bytes) -> char* {
    char* p = ws + off;
    off += (bytes + 255) & ~(size_t)255;
    return p;
  };

  bf16_t* obs_bf  = (bf16_t*)carve((size_t)M_TOT * OBS_D * 2);
  bf16_t* Wt_in   = (bf16_t*)carve((size_t)D_INNER * OBS_D * 2);      // 2048x256
  bf16_t* Wt_dt   = (bf16_t*)carve((size_t)H_HEADS * D_INNER * 2);    // 16x2048
  bf16_t* Wt_B    = (bf16_t*)carve((size_t)(H_HEADS * N_STATE) * D_INNER * 2);
  bf16_t* Wt_C    = (bf16_t*)carve((size_t)(H_HEADS * N_STATE) * D_INNER * 2);
  bf16_t* Wt_yo   = (bf16_t*)carve((size_t)N_UNITS * D_INNER * 2);    // 256x2048
  bf16_t* Wt_head = (bf16_t*)carve((size_t)N_ACT * N_UNITS * 2);      // 64x256
  bf16_t* Xbuf    = (bf16_t*)carve((size_t)M_TOT * D_INNER * 2);
  float*  dtp     = (float*) carve((size_t)M_TOT * H_HEADS * 4);
  float*  decb    = (float*) carve((size_t)M_TOT * H_HEADS * 4);
  bf16_t* Bmb     = (bf16_t*)carve((size_t)M_TOT * H_HEADS * N_STATE * 2);
  bf16_t* Cmb     = (bf16_t*)carve((size_t)M_TOT * H_HEADS * N_STATE * 2);
  bf16_t* Ybuf    = (bf16_t*)carve((size_t)M_TOT * D_INNER * 2);
  bf16_t* Zbuf    = (bf16_t*)carve((size_t)M_TOT * N_UNITS * 2);

  // --- prep: convert obs, transpose-convert weights to bf16 ---
  {
    int n = M_TOT * OBS_D;
    cvt_bf16_kernel<<<(n + 255) / 256, 256, 0, stream>>>(obs, obs_bf, n);
  }
  auto tconv = [&](const float* in, bf16_t* out, int Kd, int Nd) {
    int n = Kd * Nd;
    tconv_kernel<<<(n + 255) / 256, 256, 0, stream>>>(in, out, Kd, Nd);
  };
  tconv(W_in,   Wt_in,   OBS_D,   D_INNER);
  tconv(W_dt,   Wt_dt,   D_INNER, H_HEADS);
  tconv(W_B,    Wt_B,    D_INNER, H_HEADS * N_STATE);
  tconv(W_C,    Wt_C,    D_INNER, H_HEADS * N_STATE);
  tconv(W_yo,   Wt_yo,   D_INNER, N_UNITS);
  tconv(W_head, Wt_head, N_UNITS, N_ACT);

  // 1) X = relu(obs @ W_in + b_in)                [4096 x 2048, K=256]
  wmma_gemm_kernel<0, 4><<<dim3(M_TOT / 64, D_INNER / 128), 256, 0, stream>>>(
      obs_bf, Wt_in, b_in, nullptr, Xbuf, nullptr, M_TOT, D_INNER, OBS_D);
  // 2) dt = softplus(X @ W_dt + dt_bias); dec = exp(-exp(A_log)*dt)  [N=16]
  wmma_gemm_kernel<2, 1><<<dim3(M_TOT / 64, 1), 256, 0, stream>>>(
      Xbuf, Wt_dt, dt_bias, A_log, dtp, decb, M_TOT, H_HEADS, D_INNER);
  // 3) Bm = X @ W_B ; Cm = X @ W_C                [4096 x 1024, K=2048]
  wmma_gemm_kernel<1, 4><<<dim3(M_TOT / 64, (H_HEADS * N_STATE) / 128), 256, 0,
                           stream>>>(
      Xbuf, Wt_B, nullptr, nullptr, Bmb, nullptr, M_TOT, H_HEADS * N_STATE,
      D_INNER);
  wmma_gemm_kernel<1, 4><<<dim3(M_TOT / 64, (H_HEADS * N_STATE) / 128), 256, 0,
                           stream>>>(
      Xbuf, Wt_C, nullptr, nullptr, Cmb, nullptr, M_TOT, H_HEADS * N_STATE,
      D_INNER);
  // 4) sequential SSD scan -> Y
  ssd_scan_kernel<<<B_SZ * H_HEADS, P_HEAD, 0, stream>>>(
      Xbuf, dtp, decb, Bmb, Cmb, Ybuf);
  // 5) Z = relu(Y @ W_yo + b_yo)                  [4096 x 256, K=2048]
  wmma_gemm_kernel<3, 4><<<dim3(M_TOT / 64, N_UNITS / 128), 256, 0, stream>>>(
      Ybuf, Wt_yo, b_yo, nullptr, Zbuf, nullptr, M_TOT, N_UNITS, D_INNER);
  // 6) logits = Z @ W_head + b_head (fp32 -> d_out)  [4096 x 64, K=256]
  wmma_gemm_kernel<4, 4><<<dim3(M_TOT / 64, 1), 256, 0, stream>>>(
      Zbuf, Wt_head, b_head, nullptr, d_out, nullptr, M_TOT, N_ACT, N_UNITS);

  (void)in_sizes; (void)n_in; (void)out_size; (void)ws_size;
}
